// DenseBlockEnd_89653147337356
// MI455X (gfx1250) — matrix operand order repaired and checked by reading, not compiled, not run on hardware
//
#include <hip/hip_runtime.h>
#include <hip/hip_bf16.h>

// ---------------------------------------------------------------------------
// DenseBlockEnd fused multi-GEMM + residual + mask + ReLU for gfx1250 (MI455X)
//
//   out = relu(mask * (node + b1*sum_l(Ain_l @ Win_l) + b2*(Aout @ Wout)))
//
// Memory-bound (~0.8 GB @ 23.3 TB/s ~ 35us): bf16 WMMA keeps compute far
// below the HBM roofline. Betas folded into bf16 weights by a prologue
// swizzle that lays W out in exact B-fragment order. Weights staged to LDS
// (64KB chunks) and shared by 8 waves. Each wave owns a 16(M)x256(N) stripe
// so every activation byte is fetched from HBM exactly once, via true
// global_load (AS1) non-temporal b128 loads.
// ---------------------------------------------------------------------------

typedef __attribute__((ext_vector_type(16))) __bf16 v16bf;
typedef __attribute__((ext_vector_type(8)))  float  v8f;
typedef __attribute__((ext_vector_type(4)))  float  v4f;
typedef __attribute__((ext_vector_type(4)))  unsigned int v4u;

#define GLB __attribute__((address_space(1)))

#define BSZ      512
#define MAXA     256
#define FDIM     256
#define NLAYER   3
#define ROWS_TOT (BSZ * MAXA)          // 131072

// swizzled-W layout: [layer][kstep s:8][ntile t:16][chunk c:2][lane:32][j:8] bf16
//   per layer: 131072 B; per kstep: 16384 B; per ntile: 1024 B; per chunk: 512 B
#define WSW_LAYER_ELEMS 65536
#define WSW_TOTAL_ELEMS (NLAYER * WSW_LAYER_ELEMS)   // 196608 bf16 = 384 KB

// -------------------- prologue: scale + bf16 + fragment-swizzle W -----------
__global__ void swizzle_weights(const float* __restrict__ Win,
                                const float* __restrict__ Wout,
                                const float* __restrict__ beta1,
                                const float* __restrict__ beta2,
                                __bf16* __restrict__ dst) {
    int idx = blockIdx.x * blockDim.x + threadIdx.x;     // 0 .. 196607
    if (idx >= WSW_TOTAL_ELEMS) return;
    int l    = idx >> 16;
    int j    = idx & 7;
    int lane = (idx >> 3) & 31;
    int c    = (idx >> 8) & 1;
    int t    = (idx >> 9) & 15;
    int s    = (idx >> 13) & 7;

    int n = t * 16 + (lane & 15);
    int k = s * 32 + ((lane >> 4) << 3) + c * 16 + j;    // B frag: lanes 0-15 K=k0..7/16..23, 16-31 +8

    const float* W = (l < 2) ? (Win + (size_t)l * FDIM * FDIM) : Wout;
    float beta = (l < 2) ? beta1[0] : beta2[0];
    dst[idx] = (__bf16)(beta * W[(size_t)k * FDIM + n]);
}

// -------------------- main fused kernel -------------------------------------
// 256 threads = 8 waves; each wave: 16 rows x 256 cols; block: 128 rows.
#define LDS_CHUNK_BYTES 65536          // 4 k-steps of swizzled W

__launch_bounds__(256)
__global__ void dense_block_end(const float* __restrict__ node,
                                const float* __restrict__ inblock,
                                const float* __restrict__ bout,
                                const int*   __restrict__ mol_slice,
                                const __bf16* __restrict__ wsw,
                                float* __restrict__ out) {
    __shared__ __align__(16) unsigned char ldsW[LDS_CHUNK_BYTES];

    const int tid    = threadIdx.x;
    const int wave   = tid >> 5;                   // wave32
    const int lane   = tid & 31;
    const int laneHi = lane >> 4;                  // 0/1
    const int lane15 = lane & 15;
    const int row0   = blockIdx.x * 128 + wave * 16;

    v8f acc[16];
#pragma unroll
    for (int t = 0; t < 16; ++t) acc[t] = (v8f)0.f;

    const int aRow = row0 + lane15;                // A row handled by this lane

    for (int l = 0; l < NLAYER; ++l) {
        // uniform select, then force true global (AS1) addressing
        const float* Abase = (l == 0) ? inblock
                           : (l == 1) ? (inblock + (size_t)BSZ * MAXA * FDIM)
                                      : bout;
        GLB const float* A =
            (GLB const float*)(unsigned long long)(Abase + (size_t)aRow * FDIM);

        for (int h = 0; h < 2; ++h) {              // two 64KB halves per layer
            __syncthreads();
            {   // cooperative global->LDS copy of 4 k-steps of swizzled W (RT temporal)
                GLB const v4u* src = (GLB const v4u*)(unsigned long long)
                    (wsw + (size_t)l * WSW_LAYER_ELEMS + h * (WSW_LAYER_ELEMS / 2));
                v4u* dst = (v4u*)ldsW;
#pragma unroll
                for (int i = 0; i < LDS_CHUNK_BYTES / 16 / 256; ++i)
                    dst[tid + i * 256] = src[tid + i * 256];
            }
            __syncthreads();

#pragma unroll
            for (int s4 = 0; s4 < 4; ++s4) {
                const int k0 = (h * 4 + s4) * 32;
                const int kb = k0 + laneHi * 8;    // A frag: lane<16 K=k0..7/16..23; lane>=16 +8
                v4f a0 = __builtin_nontemporal_load((GLB const v4f*)(A + kb));
                v4f a1 = __builtin_nontemporal_load((GLB const v4f*)(A + kb + 4));
                v4f a2 = __builtin_nontemporal_load((GLB const v4f*)(A + kb + 16));
                v4f a3 = __builtin_nontemporal_load((GLB const v4f*)(A + kb + 20));
                v16bf af;
                af[0]  = (__bf16)a0.x; af[1]  = (__bf16)a0.y;
                af[2]  = (__bf16)a0.z; af[3]  = (__bf16)a0.w;
                af[4]  = (__bf16)a1.x; af[5]  = (__bf16)a1.y;
                af[6]  = (__bf16)a1.z; af[7]  = (__bf16)a1.w;
                af[8]  = (__bf16)a2.x; af[9]  = (__bf16)a2.y;
                af[10] = (__bf16)a2.z; af[11] = (__bf16)a2.w;
                af[12] = (__bf16)a3.x; af[13] = (__bf16)a3.y;
                af[14] = (__bf16)a3.z; af[15] = (__bf16)a3.w;

                const unsigned char* wbase = ldsW + s4 * 16384;
#pragma unroll
                for (int t = 0; t < 16; ++t) {
                    union { v4u u[2]; v16bf v; } bf;
                    bf.u[0] = *(const v4u*)(wbase + t * 1024 +       lane * 16);
                    bf.u[1] = *(const v4u*)(wbase + t * 1024 + 512 + lane * 16);
                    acc[t] = __builtin_amdgcn_wmma_f32_16x16x32_bf16(
                        false, af, false, bf.v, (short)0, acc[t], false, false);
                }
            }
        }
    }

    // ---------------- epilogue: node add + row mask + ReLU ------------------
    const int msv   = mol_slice[row0 >> 8];        // one molecule per 16-row tile
    const int mBase = row0 & 255;
    GLB const float* nodeg = (GLB const float*)(unsigned long long)node;
    GLB float*       outg  = (GLB float*)(unsigned long long)out;
#pragma unroll
    for (int t = 0; t < 16; ++t) {
        const int n = t * 16 + lane15;
#pragma unroll
        for (int r = 0; r < 8; ++r) {
            const int rowLocal = mBase + r + laneHi * 8;   // m within molecule
            const size_t off = (size_t)(row0 + r + laneHi * 8) * FDIM + n;
            float v = __builtin_nontemporal_load(nodeg + off) + acc[t][r];
            float res = (rowLocal < msv) ? fmaxf(v, 0.f) : 0.f;
            __builtin_nontemporal_store(res, outg + off);
        }
    }
}

// -------------------- launch -------------------------------------------------
extern "C" void kernel_launch(void* const* d_in, const int* in_sizes, int n_in,
                              void* d_out, int out_size, void* d_ws, size_t ws_size,
                              hipStream_t stream) {
    const float* node    = (const float*)d_in[0];
    const float* inblock = (const float*)d_in[1];
    const float* boutp   = (const float*)d_in[2];
    const int*   mol     = (const int*)d_in[3];
    const float* Win     = (const float*)d_in[4];
    const float* Wout    = (const float*)d_in[5];
    const float* beta1   = (const float*)d_in[6];
    const float* beta2   = (const float*)d_in[7];
    __bf16* wsw = (__bf16*)d_ws;                   // 384 KB of scratch used
    float*  out = (float*)d_out;

    swizzle_weights<<<WSW_TOTAL_ELEMS / 256, 256, 0, stream>>>(
        Win, Wout, beta1, beta2, wsw);

    dense_block_end<<<ROWS_TOT / 128, 256, 0, stream>>>(
        node, inblock, boutp, mol, wsw, out);
}
